// GavinMultiHeadPerformerAttention_20607253086355
// MI455X (gfx1250) — matrix operand order repaired and checked by reading, not compiled, not run on hardware
//
#include <hip/hip_runtime.h>

typedef __bf16 bf16;
typedef __attribute__((ext_vector_type(16))) __bf16 bf16x16;
typedef __attribute__((ext_vector_type(8)))  __bf16 bf16x8;
typedef __attribute__((ext_vector_type(2)))  __bf16 bf16x2;
typedef __attribute__((ext_vector_type(8)))  float  floatx8;

#define DEV __device__ __forceinline__

// ---- problem constants (match reference) ----
static constexpr int kB  = 4;
static constexpr int kL  = 4096;
static constexpr int kDM = 1024;
static constexpr int kH  = 16;
static constexpr int kMF = 256;           // random features
static constexpr int kBH = kB * kH;       // 64
static constexpr int kML = kB * kL;       // 16384 rows
static constexpr int kKVW = 80;           // kv ext width: 64 d + 1 ones-col + 15 pad

DEV floatx8 wmma_bf16(bf16x16 a, bf16x16 b, floatx8 c) {
  return __builtin_amdgcn_wmma_f32_16x16x32_bf16(false, a, false, b, (short)0, c,
                                                 false, false);
}

DEV floatx8 zero8() {
  floatx8 z;
#pragma unroll
  for (int i = 0; i < 8; ++i) z[i] = 0.f;
  return z;
}

// A fragment (16x32 bf16) from a row-major LDS tile base[row*stride + k].
// Per ISA 7.12.2 each lane holds two contiguous 8-element K-runs:
// lanes 0-15 (M=row): K = kbase+{0..7, 16..23}; lanes 16-31: +8.
// stride must be a multiple of 8 elements (16B) -> 2x ds_load_b128.
DEV bf16x16 frag_a_row(const bf16* base, int stride, int row, int kbase, int lane) {
  const bf16* p = base + row * stride + kbase + ((lane >> 4) ? 8 : 0);
  bf16x8 lo = *(const bf16x8*)p;         // K = kh + 0..7
  bf16x8 hi = *(const bf16x8*)(p + 16);  // K = kh + 16..23
  bf16x16 a;
#pragma unroll
  for (int j = 0; j < 8; ++j) { a[j] = lo[j]; a[8 + j] = hi[j]; }
  return a;
}

// B fragment (32x16 bf16) from an N-major LDS tile base[col*stride + k].
// Lane holds two 8-element K-runs: lanes 0-15 (N=col): K=kbase+{0..7, 16..23},
// lanes 16-31: +8 -> 2x ds_load_b128.
DEV bf16x16 frag_b_nm(const bf16* base, int stride, int col, int kbase, int lane) {
  const bf16* p = base + col * stride + kbase + ((lane >> 4) ? 8 : 0);
  bf16x8 lo = *(const bf16x8*)p;
  bf16x8 hi = *(const bf16x8*)(p + 16);
  bf16x16 b;
#pragma unroll
  for (int j = 0; j < 8; ++j) { b[j] = lo[j]; b[8 + j] = hi[j]; }
  return b;
}

DEV void store_pair(bf16* p, float a, float b) {  // one ds_store_b32
  bf16x2 v; v[0] = (bf16)a; v[1] = (bf16)b;
  *(bf16x2*)p = v;
}
DEV void store_pair_bf(bf16* p, bf16x2 v) { *(bf16x2*)p = v; }

DEV bf16 as_bf(float x) { return (bf16)x; }
DEV bf16 as_bf(bf16 x)  { return x; }
DEV void store_out(float* p, float v) { *p = v; }
DEV void store_out(bf16* p, float v)  { *p = (bf16)v; }

DEV void atomicMaxFloat(float* addr, float val) {
  unsigned int* ua = (unsigned int*)addr;
  unsigned int old = __float_as_uint(*addr);
  while (true) {
    float f = __uint_as_float(old);
    if (f >= val) break;
    unsigned int assumed = old;
    old = atomicCAS(ua, assumed, __float_as_uint(val));
    if (old == assumed) break;
  }
}

// ======================================================================
// Generic GEMM + bias: C[M,N] = A[M,K] @ W[K,N] + bias.
// Block tile 64x128, 8 waves, wave = 16 rows x 64 cols (4 WMMA accs).
// As row-major [64][40]; Bs N-major [128][40] (transposed at fill time).
// ======================================================================
template <typename AT, typename OT>
__global__ __launch_bounds__(256) void gemm_bias_kernel(
    const AT* __restrict__ A, const float* __restrict__ W,
    const float* __restrict__ bias, OT* __restrict__ C, int M, int N, int K) {
  __shared__ alignas(16) bf16 As[64][40];
  __shared__ alignas(16) bf16 Bs[128][40];  // [col][k]
  const int tid = threadIdx.x, lane = tid & 31, w = tid >> 5;
  const int mblk = blockIdx.y * 64, nblk = blockIdx.x * 128;
  const int wm = (w & 3) * 16;
  const int wn = (w >> 2) * 64;

  floatx8 acc[4];
#pragma unroll
  for (int t = 0; t < 4; ++t) acc[t] = zero8();

  for (int k0 = 0; k0 < K; k0 += 32) {
    // A tile 64x32 (row-major), packed-pair stores
    for (int i = tid; i < 64 * 16; i += 256) {
      int r = i >> 4, cp = (i & 15) * 2;
      float a0 = (float)A[(size_t)(mblk + r) * K + k0 + cp];
      float a1 = (float)A[(size_t)(mblk + r) * K + k0 + cp + 1];
      store_pair(&As[r][cp], a0, a1);
    }
    // W tile 32x128 -> N-major Bs[col][k]
    for (int i = tid; i < 16 * 128; i += 256) {
      int kp = (i >> 7) * 2, c = i & 127;
      float w0 = W[(size_t)(k0 + kp) * N + nblk + c];
      float w1 = W[(size_t)(k0 + kp + 1) * N + nblk + c];
      store_pair(&Bs[c][kp], w0, w1);
    }
    if (k0 + 32 < K) {  // gfx1250 global_prefetch_b8 for next K tile
      __builtin_prefetch(&A[(size_t)(mblk + (tid & 63)) * K + k0 + 32], 0, 1);
      __builtin_prefetch(&W[(size_t)(k0 + 32 + (tid & 31)) * N + nblk], 0, 1);
    }
    __syncthreads();
    bf16x16 af = frag_a_row(&As[0][0], 40, wm + (lane & 15), 0, lane);
#pragma unroll
    for (int t = 0; t < 4; ++t) {
      bf16x16 bfr = frag_b_nm(&Bs[0][0], 40, wn + t * 16 + (lane & 15), 0, lane);
      acc[t] = wmma_bf16(af, bfr, acc[t]);
    }
    __syncthreads();
  }

  const int rbase = mblk + wm + ((lane >> 4) ? 8 : 0);
#pragma unroll
  for (int t = 0; t < 4; ++t) {
    int col = nblk + wn + t * 16 + (lane & 15);
    float bv = bias ? bias[col] : 0.f;
#pragma unroll
    for (int r = 0; r < 8; ++r)
      store_out(&C[(size_t)(rbase + r) * N + col], acc[t][r] + bv);
  }
}

// ======================================================================
// FAVOR+ feature map.  data_dash = (x * d^-0.25) @ proj^T (scale folded
// into the P tile).  mode 0: global key max per (b,h) (atomicMax).
// mode 1: query (row max) -> q'.   mode 2: key (uses kmax) -> k'.
// Tile: 32 L-rows x 256 features, K = D = 64 (two WMMA K-steps).
// ======================================================================
__global__ __launch_bounds__(256) void feat_kernel(
    const bf16* __restrict__ inp, const float* __restrict__ feats,
    bf16* __restrict__ outp, float* __restrict__ kmax, int mode) {
  __shared__ alignas(16) bf16 As[32][72];    // [l][d] row-major
  __shared__ alignas(16) bf16 Bs[256][72];   // [m][d] N-major
  __shared__ alignas(16) bf16 Cs[32][258];   // data_dash staging
  __shared__ float red[256];
  __shared__ float rowred[32][8];
  __shared__ float rowsq[32][8];
  __shared__ float rowmax[32];
  __shared__ float rowdiag[32];

  const int tid = threadIdx.x, lane = tid & 31, w = tid >> 5;
  const int bh = blockIdx.x, b = bh >> 4, h = bh & 15;
  const int lbase = blockIdx.y * 32;
  const float dscale = 0.35355339059327379f;  // 64^-0.25

  // B = (proj * scale), N-major: Bs[m][d]  (m-row of feats is contiguous)
  for (int i = tid; i < 256 * 32; i += 256) {
    int m = i >> 5, dp = (i & 31) * 2;
    store_pair(&Bs[m][dp], feats[m * 64 + dp] * dscale, feats[m * 64 + dp + 1] * dscale);
  }
  // A = projected rows for this head (raw, unscaled): 32 x 64
  for (int i = tid; i < 32 * 32; i += 256) {
    int r = i >> 5, dp = (i & 31) * 2;
    bf16x2 v = *(const bf16x2*)&inp[(size_t)(b * kL + lbase + r) * kDM + h * 64 + dp];
    store_pair_bf(&As[r][dp], v);
  }
  __syncthreads();

  const int rt = (w & 1) * 16;
  const int cg = (w >> 1) * 64;
  floatx8 acc[4];
#pragma unroll
  for (int t = 0; t < 4; ++t) acc[t] = zero8();
#pragma unroll
  for (int ks = 0; ks < 2; ++ks) {
    bf16x16 af = frag_a_row(&As[0][0], 72, rt + (lane & 15), ks * 32, lane);
#pragma unroll
    for (int t = 0; t < 4; ++t) {
      bf16x16 bfr = frag_b_nm(&Bs[0][0], 72, cg + t * 16 + (lane & 15), ks * 32, lane);
      acc[t] = wmma_bf16(af, bfr, acc[t]);
    }
  }
  {  // stage data_dash tiles to LDS (C layout: M=r / M=r+8 lane halves)
    int r0 = rt + ((lane >> 4) ? 8 : 0);
    int c0 = cg + (lane & 15);
#pragma unroll
    for (int t = 0; t < 4; ++t)
#pragma unroll
      for (int r = 0; r < 8; ++r) Cs[r0 + r][c0 + t * 16] = (bf16)acc[t][r];
  }
  __syncthreads();

  if (mode == 0) {  // key max pre-pass: global max over (L, M) per (b,h)
    float m = -3.0e38f;
    for (int i = tid; i < 32 * 256; i += 256) {
      int r = i >> 8, c = i & 255;
      m = fmaxf(m, (float)Cs[r][c]);
    }
    red[tid] = m;
    __syncthreads();
    for (int s = 128; s > 0; s >>= 1) {
      if (tid < s) red[tid] = fmaxf(red[tid], red[tid + s]);
      __syncthreads();
    }
    if (tid == 0) atomicMaxFloat(&kmax[bh], red[0]);
    return;
  }

  // exp phase: 8 threads per row (32 rows x 8 = 256)
  const int row = tid >> 3, s = tid & 7;
  float pm = -3.0e38f;
  if (mode == 1) {
    for (int c = s * 32; c < s * 32 + 32; ++c) pm = fmaxf(pm, (float)Cs[row][c]);
  }
  float qs = 0.f;
  for (int d = s * 8; d < s * 8 + 8; ++d) {
    float qv = (float)As[row][d];
    qs += qv * qv;
  }
  rowred[row][s] = pm;
  rowsq[row][s]  = qs;
  __syncthreads();
  if (s == 0) {
    float mx;
    if (mode == 1) {
      mx = rowred[row][0];
      for (int j = 1; j < 8; ++j) mx = fmaxf(mx, rowred[row][j]);
    } else {
      mx = kmax[bh];
    }
    float dsum = 0.f;
    for (int j = 0; j < 8; ++j) dsum += rowsq[row][j];
    rowmax[row]  = mx;
    rowdiag[row] = 0.0625f * dsum;  // 0.5 * 64^-0.5 * sum(raw^2)
  }
  __syncthreads();
  const float mx = rowmax[row], dg = rowdiag[row];
  const size_t obase = ((size_t)bh * kL + lbase + row) * kMF;
  for (int c = s * 32; c < s * 32 + 32; ++c) {
    float v = 0.0625f * (expf((float)Cs[row][c] - dg - mx) + 1e-6f);  // ratio=M^-0.5
    outp[obase + c] = (bf16)v;
  }
}

// ======================================================================
// kv_ext[bh][256][80] = k'^T[256 x L] @ v_ext[L x 80], v_ext col 64 = 1
// (column 64 of the output is sum_L k' — the attention normalizer).
// One block per (b,h); 8 waves; wave = 32 m-rows x 80 cols (10 accs);
// K = L = 4096 stepped by 32.  Ks stored m-major so k'^T A-frags are
// contiguous; Vs stored N-major.
// ======================================================================
__global__ __launch_bounds__(256) void kv_kernel(const bf16* __restrict__ kp,
                                                 const bf16* __restrict__ vproj,
                                                 float* __restrict__ kv) {
  __shared__ alignas(16) bf16 Ks[256][40];  // [m][l_local]
  __shared__ alignas(16) bf16 Vs[80][40];   // [d_ext][l_local]
  const int tid = threadIdx.x, lane = tid & 31, w = tid >> 5;
  const int bh = blockIdx.x, b = bh >> 4, h = bh & 15;

  floatx8 acc[2][5];
#pragma unroll
  for (int i = 0; i < 2; ++i)
#pragma unroll
    for (int j = 0; j < 5; ++j) acc[i][j] = zero8();

  for (int l0 = 0; l0 < kL; l0 += 32) {
    // k' tile: global [l][m] -> LDS [m][l] (coalesced reads along m)
    for (int i = tid; i < 16 * 256; i += 256) {
      int m = i & 255, lp = (i >> 8) * 2;
      float k0v = (float)kp[((size_t)bh * kL + l0 + lp) * kMF + m];
      float k1v = (float)kp[((size_t)bh * kL + l0 + lp + 1) * kMF + m];
      store_pair(&Ks[m][lp], k0v, k1v);
    }
    // v_ext tile: global [l][d] -> LDS [d_ext][l]; col 64 = ones
    for (int i = tid; i < 80 * 16; i += 256) {
      int c = i % 80, lp = (i / 80) * 2;
      float v0, v1;
      if (c < 64) {
        v0 = (float)vproj[(size_t)(b * kL + l0 + lp) * kDM + h * 64 + c];
        v1 = (float)vproj[(size_t)(b * kL + l0 + lp + 1) * kDM + h * 64 + c];
      } else if (c == 64) { v0 = 1.f; v1 = 1.f; }
      else                { v0 = 0.f; v1 = 0.f; }
      store_pair(&Vs[c][lp], v0, v1);
    }
    __syncthreads();
    bf16x16 bfr[5];
#pragma unroll
    for (int nt = 0; nt < 5; ++nt)
      bfr[nt] = frag_b_nm(&Vs[0][0], 40, nt * 16 + (lane & 15), 0, lane);
#pragma unroll
    for (int mt = 0; mt < 2; ++mt) {
      bf16x16 af = frag_a_row(&Ks[0][0], 40, (2 * w + mt) * 16 + (lane & 15), 0, lane);
#pragma unroll
      for (int nt = 0; nt < 5; ++nt) acc[mt][nt] = wmma_bf16(af, bfr[nt], acc[mt][nt]);
    }
    __syncthreads();
  }

#pragma unroll
  for (int mt = 0; mt < 2; ++mt) {
    int r0 = (2 * w + mt) * 16 + ((lane >> 4) ? 8 : 0);
#pragma unroll
    for (int nt = 0; nt < 5; ++nt) {
      int col = nt * 16 + (lane & 15);
#pragma unroll
      for (int r = 0; r < 8; ++r)
        kv[(size_t)bh * kMF * kKVW + (size_t)(r0 + r) * kKVW + col] = acc[mt][nt][r];
    }
  }
}

// ======================================================================
// av: [128 x 80] = q'[128 x 256] @ kv_ext[256 x 80]; column 64 is the
// normalizer; divide and write head-interleaved bf16 attn[B*L, DM].
// Qs row-major; KVs N-major [80][264].
// ======================================================================
__global__ __launch_bounds__(256) void av_kernel(const bf16* __restrict__ qp,
                                                 const float* __restrict__ kv,
                                                 bf16* __restrict__ attn) {
  __shared__ alignas(16) bf16  Qs[128][40];
  __shared__ alignas(16) bf16  KVs[80][264];  // [d_ext][m]
  __shared__ float normbuf[128];
  const int tid = threadIdx.x, lane = tid & 31, w = tid >> 5;
  const int bh = blockIdx.x, b = bh >> 4, h = bh & 15;
  const int lbase = blockIdx.y * 128;
  const int wrow = w * 16;

  // kv_ext: global [m][d] f32 -> LDS [d][m] bf16 (coalesced reads along d)
  for (int i = tid; i < 80 * 128; i += 256) {
    int c = i % 80, mp = (i / 80) * 2;
    float v0 = kv[(size_t)bh * kMF * kKVW + (size_t)mp * kKVW + c];
    float v1 = kv[(size_t)bh * kMF * kKVW + (size_t)(mp + 1) * kKVW + c];
    store_pair(&KVs[c][mp], v0, v1);
  }

  floatx8 acc[5];
#pragma unroll
  for (int t = 0; t < 5; ++t) acc[t] = zero8();

  for (int kb0 = 0; kb0 < kMF; kb0 += 32) {
    for (int i = tid; i < 128 * 16; i += 256) {
      int r = i >> 4, cp = (i & 15) * 2;
      bf16x2 v = *(const bf16x2*)&qp[((size_t)bh * kL + lbase + r) * kMF + kb0 + cp];
      store_pair_bf(&Qs[r][cp], v);
    }
    __syncthreads();
    bf16x16 af = frag_a_row(&Qs[0][0], 40, wrow + (lane & 15), 0, lane);
#pragma unroll
    for (int nt = 0; nt < 5; ++nt) {
      bf16x16 bfr = frag_b_nm(&KVs[0][0], 264, nt * 16 + (lane & 15), kb0, lane);
      acc[nt] = wmma_bf16(af, bfr, acc[nt]);
    }
    __syncthreads();
  }

  // acc[4] column 0 (global col 64) is the norm: lanes 0 & 16 own it.
  const int r0 = wrow + ((lane >> 4) ? 8 : 0);
  if ((lane & 15) == 0) {
#pragma unroll
    for (int r = 0; r < 8; ++r) normbuf[r0 + r] = acc[4][r];
  }
  __syncthreads();
#pragma unroll
  for (int nt = 0; nt < 4; ++nt) {
    int col = nt * 16 + (lane & 15);
#pragma unroll
    for (int r = 0; r < 8; ++r) {
      int row = r0 + r;
      float v = acc[nt][r] / normbuf[row];
      attn[(size_t)(b * kL + lbase + row) * kDM + h * 64 + col] = (bf16)v;
    }
  }
}

__global__ void init_kmax_kernel(float* kmax) {
  if (threadIdx.x < kBH) kmax[threadIdx.x] = -3.0e38f;
}

// ======================================================================
extern "C" void kernel_launch(void* const* d_in, const int* in_sizes, int n_in,
                              void* d_out, int out_size, void* d_ws, size_t ws_size,
                              hipStream_t stream) {
  (void)in_sizes; (void)n_in; (void)out_size; (void)ws_size;
  const float* q_in  = (const float*)d_in[0];
  const float* k_in  = (const float*)d_in[1];
  const float* v_in  = (const float*)d_in[2];
  const float* Wq    = (const float*)d_in[3];
  const float* bq    = (const float*)d_in[4];
  const float* Wk    = (const float*)d_in[5];
  const float* bk    = (const float*)d_in[6];
  const float* Wv    = (const float*)d_in[7];
  const float* bv    = (const float*)d_in[8];
  const float* Wo    = (const float*)d_in[9];
  const float* bo    = (const float*)d_in[10];
  const float* feats = (const float*)d_in[11];

  char* ws = (char*)d_ws;
  const size_t MB = 1ull << 20;
  bf16*  qproj = (bf16*)(ws + 0 * MB);     // 32 MB  [B*L, DM] bf16
  bf16*  kproj = (bf16*)(ws + 32 * MB);    // 32 MB
  bf16*  vproj = (bf16*)(ws + 64 * MB);    // 32 MB
  bf16*  qp    = (bf16*)(ws + 96 * MB);    // 128 MB [BH, L, 256] bf16
  bf16*  kp    = (bf16*)(ws + 224 * MB);   // 128 MB
  float* kvb   = (float*)(ws + 352 * MB);  // 5 MB   [BH, 256, 80] f32
  bf16*  attn  = (bf16*)(ws + 360 * MB);   // 32 MB  [B*L, DM] bf16
  float* kmax  = (float*)(ws + 392 * MB);  // 256 B  [BH] f32

  init_kmax_kernel<<<1, kBH, 0, stream>>>(kmax);

  dim3 ggrid(kDM / 128, kML / 64);  // (8, 256)
  gemm_bias_kernel<float, bf16><<<ggrid, 256, 0, stream>>>(q_in, Wq, bq, qproj, kML, kDM, kDM);
  gemm_bias_kernel<float, bf16><<<ggrid, 256, 0, stream>>>(k_in, Wk, bk, kproj, kML, kDM, kDM);
  gemm_bias_kernel<float, bf16><<<ggrid, 256, 0, stream>>>(v_in, Wv, bv, vproj, kML, kDM, kDM);

  dim3 fgrid(kBH, kL / 32);  // (64, 128)
  feat_kernel<<<fgrid, 256, 0, stream>>>(kproj, feats, (bf16*)nullptr, kmax, 0);  // key max
  feat_kernel<<<fgrid, 256, 0, stream>>>(kproj, feats, kp, kmax, 2);              // k'
  feat_kernel<<<fgrid, 256, 0, stream>>>(qproj, feats, qp, (float*)nullptr, 1);   // q'

  kv_kernel<<<kBH, 256, 0, stream>>>(kp, vproj, kvb);
  av_kernel<<<dim3(kBH, kL / 128), 256, 0, stream>>>(qp, kvb, attn);

  gemm_bias_kernel<bf16, float><<<ggrid, 256, 0, stream>>>(attn, Wo, bo, (float*)d_out, kML, kDM, kDM);
}
